// FinalCompactModel_47863115547155
// MI455X (gfx1250) — matrix operand order repaired and checked by reading, not compiled, not run on hardware
//
#include <hip/hip_runtime.h>
#include <hip/hip_bf16.h>
#include <math.h>

// ---------------------------------------------------------------------------
// Types for CDNA5 WMMA (wave32, v_wmma_f32_16x16x32_bf16)
// ---------------------------------------------------------------------------
typedef __attribute__((ext_vector_type(16))) __bf16 v16bf;
typedef __attribute__((ext_vector_type(8)))  __bf16 v8bf;
typedef __attribute__((ext_vector_type(8)))  float  v8f;

#define B_  4096
#define S_  60
#define L_  20000
#define D_  96
#define H_  4
#define DH_ 24
#define BS_ (B_ * S_)

__device__ __forceinline__ float gelu_exact(float x) {
    return 0.5f * x * (1.0f + erff(x * 0.70710678118654752f));
}

// ---------------------------------------------------------------------------
// f32 -> bf16 conversion (weights + activations shadow copies)
// ---------------------------------------------------------------------------
__global__ void cvt_f32_bf16_kernel(const float* __restrict__ src,
                                    __bf16* __restrict__ dst, int n) {
    int i = blockIdx.x * 256 + threadIdx.x;
    if (i < n) dst[i] = (__bf16)src[i];
}

// ---------------------------------------------------------------------------
// Embedding gather + temporal MLP + LayerNorm + positional encoding.
// One wave per token, 3 channels per lane (32*3 = 96 = D).
// ---------------------------------------------------------------------------
__global__ __launch_bounds__(256)
void embed_kernel(const int* __restrict__ loc_seq, const int* __restrict__ user_seq,
                  const int* __restrict__ weekday_seq,
                  const float* __restrict__ start_min, const float* __restrict__ dur,
                  const float* __restrict__ diff,
                  const float* __restrict__ loc_w, const float* __restrict__ user_w,
                  const float* __restrict__ tw, const float* __restrict__ tb,
                  const float* __restrict__ g, const float* __restrict__ beta,
                  const float* __restrict__ pe,
                  float* __restrict__ xf, __bf16* __restrict__ xb, int ntok) {
    const int lane = threadIdx.x & 31;
    const int wave = threadIdx.x >> 5;
    const int tok  = blockIdx.x * 8 + wave;
    if (tok >= ntok) return;
    const int s   = tok % S_;
    const int loc = loc_seq[tok];
    const int usr = user_seq[tok];

    float feats[6];
    {
        const float two_pi = 6.28318530717958648f;
        float hours = start_min[tok] * (1.0f / 60.0f);
        float tr = hours * (two_pi / 24.0f);
        float wr = (float)weekday_seq[tok] * (two_pi / 7.0f);
        feats[0] = __sinf(tr);
        feats[1] = __cosf(tr);
        feats[2] = log1pf(dur[tok]) * 0.125f;
        feats[3] = __sinf(wr);
        feats[4] = __cosf(wr);
        feats[5] = diff[tok] * (1.0f / 7.0f);
    }

    float v[3];
#pragma unroll
    for (int j = 0; j < 3; ++j) {
        int d = lane * 3 + j;
        float val;
        if (d < 64) {
            val = loc_w[(size_t)loc * 64 + d];
        } else if (d < 80) {
            val = user_w[(size_t)usr * 16 + (d - 64)];
        } else {
            int o = d - 80;
            float a = tb[o];
#pragma unroll
            for (int q = 0; q < 6; ++q) a += tw[o * 6 + q] * feats[q];
            val = a;
        }
        v[j] = val;
    }

    float sum = v[0] + v[1] + v[2];
#pragma unroll
    for (int off = 16; off > 0; off >>= 1) sum += __shfl_xor(sum, off, 32);
    float mean = sum * (1.0f / 96.0f);
    float var = 0.f;
#pragma unroll
    for (int j = 0; j < 3; ++j) { float t = v[j] - mean; var += t * t; }
#pragma unroll
    for (int off = 16; off > 0; off >>= 1) var += __shfl_xor(var, off, 32);
    float inv = rsqrtf(var * (1.0f / 96.0f) + 1e-5f);

#pragma unroll
    for (int j = 0; j < 3; ++j) {
        int d = lane * 3 + j;
        float y = (v[j] - mean) * inv * g[d] + beta[d] + pe[s * D_ + d];
        size_t off = (size_t)tok * D_ + d;
        xf[off] = y;
        xb[off] = (__bf16)y;
    }
}

// ---------------------------------------------------------------------------
// Residual + LayerNorm, one wave per token (in-place capable).
// ---------------------------------------------------------------------------
__global__ __launch_bounds__(256)
void ln_res_kernel(const float* __restrict__ x, const float* __restrict__ r,
                   const float* __restrict__ g, const float* __restrict__ beta,
                   float* __restrict__ yf, __bf16* __restrict__ yb, int ntok) {
    const int lane = threadIdx.x & 31;
    const int wave = threadIdx.x >> 5;
    const int tok  = blockIdx.x * 8 + wave;
    if (tok >= ntok) return;

    float v[3];
#pragma unroll
    for (int j = 0; j < 3; ++j) {
        int d = lane * 3 + j;
        size_t off = (size_t)tok * D_ + d;
        float val = x[off];
        if (r) val += r[off];
        v[j] = val;
    }
    float sum = v[0] + v[1] + v[2];
#pragma unroll
    for (int off = 16; off > 0; off >>= 1) sum += __shfl_xor(sum, off, 32);
    float mean = sum * (1.0f / 96.0f);
    float var = 0.f;
#pragma unroll
    for (int j = 0; j < 3; ++j) { float t = v[j] - mean; var += t * t; }
#pragma unroll
    for (int off = 16; off > 0; off >>= 1) var += __shfl_xor(var, off, 32);
    float inv = rsqrtf(var * (1.0f / 96.0f) + 1e-5f);

#pragma unroll
    for (int j = 0; j < 3; ++j) {
        int d = lane * 3 + j;
        float y = (v[j] - mean) * inv * g[d] + beta[d];
        size_t off = (size_t)tok * D_ + d;
        if (yf) yf[off] = y;
        if (yb) yb[off] = (__bf16)y;
    }
}

// ---------------------------------------------------------------------------
// Generic bf16 WMMA GEMM:  C[M,N] = act(A[M,K] * Wt[N,K]^T + bias)
// Wave tile: 32(M) x 64(N) -> 8 WMMAs per K-step sharing 4 B fragments.
// 8 waves/block -> 256 x 64 block tile. K templated (96/192): full unroll,
// straight-line inner loop (no branches; EXEC all-1s around every WMMA).
// Out-of-range sub-tile columns are clamped for loads, predicated for stores.
// Requires: M % 32 == 0 (grid covers M in 256-row blocks), N >= 64,
//           N % 16 == 0, K % 32 == 0.
// NT=1: non-temporal stores for the huge streamed output (ISA TH=NT).
// ---------------------------------------------------------------------------
template <int K, int ACT, int NT>
__global__ __launch_bounds__(256)
void gemm_bf16_wmma(const __bf16* __restrict__ A,   // [M,K] row-major
                    const __bf16* __restrict__ Wt,  // [N,K] row-major (B^T)
                    const float*  __restrict__ bias,// [N] or null
                    float* __restrict__ Cf,         // [M,N] or null
                    __bf16* __restrict__ Cb,        // [M,N] or null
                    int M, int N) {
    const int lane = threadIdx.x & 31;
    const int wave = threadIdx.x >> 5;
    const int m0    = blockIdx.y * 256 + wave * 32;
    const int n0raw = blockIdx.x * 64;
    if (m0 >= M) return;

    // ISA 7.12.2 bf16 A layout: lanes 0-15 -> K {0..7,16..23}; 16-31 -> K {8..15,24..31}
    const int mRow  = lane & 15;
    const int aKoff = (lane < 16) ? 0 : 8;
    // bf16 B layout (KxN): lanes 0-15 -> K {0..7,16..23} of col N=lane; 16-31 -> K {8..15,24..31}
    const int bKoff = (lane < 16) ? 0 : 8;
    const int nCol  = lane & 15;

    // Clamp sub-tile columns so every load/WMMA is unconditional (wave-uniform).
    int nb[4];
#pragma unroll
    for (int t = 0; t < 4; ++t) {
        int nt = n0raw + t * 16;
        nb[t] = (nt < N) ? nt : (N - 16);
    }

    const __bf16* aRow0 = A + (size_t)(m0 + mRow) * K + aKoff;
    const __bf16* aRow1 = aRow0 + (size_t)16 * K;
    const __bf16* bCol[4];
#pragma unroll
    for (int t = 0; t < 4; ++t) bCol[t] = Wt + (size_t)(nb[t] + nCol) * K + bKoff;

    v8f acc[2][4];
#pragma unroll
    for (int h = 0; h < 2; ++h)
#pragma unroll
        for (int t = 0; t < 4; ++t) acc[h][t] = v8f{};

#pragma unroll
    for (int k0 = 0; k0 < K; k0 += 32) {
        if (k0 + 32 < K) {
            __builtin_prefetch(aRow0 + k0 + 32, 0, 1);  // global_prefetch_b8
            __builtin_prefetch(aRow1 + k0 + 32, 0, 1);
        }
        v8bf a0lo = *(const v8bf*)(aRow0 + k0);
        v8bf a0hi = *(const v8bf*)(aRow0 + k0 + 16);
        v8bf a1lo = *(const v8bf*)(aRow1 + k0);
        v8bf a1hi = *(const v8bf*)(aRow1 + k0 + 16);
        v16bf a0, a1;
#pragma unroll
        for (int i = 0; i < 8; ++i) {
            a0[i] = a0lo[i]; a0[8 + i] = a0hi[i];
            a1[i] = a1lo[i]; a1[8 + i] = a1hi[i];
        }
#pragma unroll
        for (int t = 0; t < 4; ++t) {
            v8bf blo = *(const v8bf*)(bCol[t] + k0);
            v8bf bhi = *(const v8bf*)(bCol[t] + k0 + 16);
            v16bf bfrag;
#pragma unroll
            for (int i = 0; i < 8; ++i) { bfrag[i] = blo[i]; bfrag[8 + i] = bhi[i]; }
            acc[0][t] = __builtin_amdgcn_wmma_f32_16x16x32_bf16(
                false, a0, false, bfrag, (short)0, acc[0][t], false, false);
            acc[1][t] = __builtin_amdgcn_wmma_f32_16x16x32_bf16(
                false, a1, false, bfrag, (short)0, acc[1][t], false, false);
        }
    }

    // C/D layout: VGPR r, lanes 0-15 -> M=r, lanes 16-31 -> M=r+8; N = lane&15
    const int mOff = (lane < 16) ? 0 : 8;
#pragma unroll
    for (int h = 0; h < 2; ++h) {
        const int mbase = m0 + h * 16 + mOff;
#pragma unroll
        for (int t = 0; t < 4; ++t) {
            if (n0raw + t * 16 >= N) continue;  // wave-uniform store guard
            int n = nb[t] + nCol;
            float bv = bias ? bias[n] : 0.0f;
#pragma unroll
            for (int r = 0; r < 8; ++r) {
                float v = acc[h][t][r] + bv;
                if (ACT == 1) v = gelu_exact(v);
                size_t off = (size_t)(mbase + r) * N + n;
                if (Cf) {
                    if (NT) __builtin_nontemporal_store(v, &Cf[off]);
                    else    Cf[off] = v;
                }
                if (Cb) Cb[off] = (__bf16)v;
            }
        }
    }
}

// ---------------------------------------------------------------------------
// Attention: one block (64 threads) per (batch, head). K/V staged in LDS.
// S=60, DH=24. Outputs attn*V directly as bf16 for the out_proj GEMM.
// ---------------------------------------------------------------------------
__global__ __launch_bounds__(64)
void attn_kernel(const float* __restrict__ qkv, const unsigned char* __restrict__ mask,
                 __bf16* __restrict__ aob) {
    __shared__ float Ks[S_ * DH_];
    __shared__ float Vs[S_ * DH_];
    const int bh = blockIdx.x;
    const int b  = bh >> 2;
    const int h  = bh & 3;
    const float* base = qkv + (size_t)b * S_ * (3 * D_);
    const int tid = threadIdx.x;

    for (int idx = tid; idx < S_ * DH_; idx += 64) {
        int s = idx / DH_, d = idx - s * DH_;
        Ks[idx] = base[s * (3 * D_) + D_ + h * DH_ + d];
        Vs[idx] = base[s * (3 * D_) + 2 * D_ + h * DH_ + d];
    }
    __syncthreads();
    if (tid >= S_) return;

    float q[DH_];
#pragma unroll
    for (int d = 0; d < DH_; ++d) q[d] = base[tid * (3 * D_) + h * DH_ + d];

    float sc[S_];
    float mx = -1e30f;
    const float scale = 0.20412414523193150f;  // 1/sqrt(24)
    for (int k = 0; k < S_; ++k) {
        float a = 0.f;
#pragma unroll
        for (int d = 0; d < DH_; ++d) a += q[d] * Ks[k * DH_ + d];
        a *= scale;
        if (!mask[b * S_ + k]) a = -1e9f;
        sc[k] = a;
        mx = fmaxf(mx, a);
    }
    float ssum = 0.f;
    for (int k = 0; k < S_; ++k) { float e = __expf(sc[k] - mx); sc[k] = e; ssum += e; }
    float invs = 1.0f / ssum;
#pragma unroll
    for (int d = 0; d < DH_; ++d) {
        float a = 0.f;
        for (int k = 0; k < S_; ++k) a += sc[k] * Vs[k * DH_ + d];
        aob[((size_t)b * S_ + tid) * D_ + h * DH_ + d] = (__bf16)(a * invs);
    }
}

// ---------------------------------------------------------------------------
// Gather last valid hidden state -> bf16 [B, 96]
// ---------------------------------------------------------------------------
__global__ void last_kernel(const float* __restrict__ x2,
                            const unsigned char* __restrict__ mask,
                            __bf16* __restrict__ lastb) {
    int i = blockIdx.x * 256 + threadIdx.x;
    if (i >= B_ * D_) return;
    int b = i / D_, d = i - b * D_;
    int sl = 0;
    for (int s = 0; s < S_; ++s) sl += mask[b * S_ + s] ? 1 : 0;
    sl -= 1;
    lastb[i] = (__bf16)x2[((size_t)b * S_ + sl) * D_ + d];
}

// ---------------------------------------------------------------------------
// Recency-weighted history add: one thread per batch row (row-private,
// serial RMW -> deterministic, no atomics). Runs AFTER the logits GEMM.
// ---------------------------------------------------------------------------
__global__ void hist_kernel(const int* __restrict__ loc_seq,
                            const unsigned char* __restrict__ mask,
                            const float* __restrict__ boost,
                            const float* __restrict__ decay,
                            float* __restrict__ out) {
    int b = blockIdx.x * 256 + threadIdx.x;
    if (b >= B_) return;
    float hb = boost[0], rd = decay[0];
    float rec = 1.0f;  // decay^(S-1-s), s descending from S-1
    for (int s = S_ - 1; s >= 0; --s) {
        if (mask[b * S_ + s]) {
            int loc = loc_seq[b * S_ + s];
            size_t off = (size_t)b * L_ + loc;
            out[off] += hb * rec;
        }
        rec *= rd;
    }
}

// ---------------------------------------------------------------------------
// Host-side launch
// ---------------------------------------------------------------------------
template <int K, int ACT, int NT>
static inline void launch_gemm(const __bf16* A, const __bf16* Wt, const float* bias,
                               float* Cf, __bf16* Cb, int M, int N,
                               hipStream_t stream) {
    dim3 grid((N + 63) / 64, (M + 255) / 256);
    gemm_bf16_wmma<K, ACT, NT><<<grid, 256, 0, stream>>>(A, Wt, bias, Cf, Cb, M, N);
}

static inline void launch_cvt(const float* s, __bf16* d, int n, hipStream_t stream) {
    cvt_f32_bf16_kernel<<<(n + 255) / 256, 256, 0, stream>>>(s, d, n);
}

extern "C" void kernel_launch(void* const* d_in, const int* in_sizes, int n_in,
                              void* d_out, int out_size, void* d_ws, size_t ws_size,
                              hipStream_t stream) {
    (void)in_sizes; (void)n_in; (void)out_size; (void)ws_size;
    // ---- inputs (setup_inputs order) ----
    const int*   loc_seq    = (const int*)d_in[0];
    const int*   user_seq   = (const int*)d_in[1];
    const int*   weekday    = (const int*)d_in[2];
    const float* start_min  = (const float*)d_in[3];
    const float* dur        = (const float*)d_in[4];
    const float* diff       = (const float*)d_in[5];
    const unsigned char* mask = (const unsigned char*)d_in[6];
    const float* loc_emb_w  = (const float*)d_in[7];
    const float* user_emb_w = (const float*)d_in[8];
    const float* temporal_w = (const float*)d_in[9];
    const float* temporal_b = (const float*)d_in[10];
    const float* in_norm_g  = (const float*)d_in[11];
    const float* in_norm_b  = (const float*)d_in[12];
    const float* pe         = (const float*)d_in[13];
    const float* in_proj_w  = (const float*)d_in[14];
    const float* in_proj_b  = (const float*)d_in[15];
    const float* out_proj_w = (const float*)d_in[16];
    const float* out_proj_b = (const float*)d_in[17];
    const float* norm1_g    = (const float*)d_in[18];
    const float* norm1_b    = (const float*)d_in[19];
    const float* ff_w1      = (const float*)d_in[20];
    const float* ff_b1      = (const float*)d_in[21];
    const float* ff_w2      = (const float*)d_in[22];
    const float* ff_b2      = (const float*)d_in[23];
    const float* norm2_g    = (const float*)d_in[24];
    const float* norm2_b    = (const float*)d_in[25];
    const float* pred_w1    = (const float*)d_in[26];
    const float* pred_b1    = (const float*)d_in[27];
    const float* pred_w2    = (const float*)d_in[28];
    const float* pred_b2    = (const float*)d_in[29];
    const float* boost      = (const float*)d_in[30];
    const float* decay      = (const float*)d_in[31];
    float* out = (float*)d_out;

    // ---- workspace layout (all offsets 256B aligned) ----
    char* ws = (char*)d_ws;
    const size_t XF   = (size_t)BS_ * D_ * 4;       // 94,371,840
    const size_t XB   = (size_t)BS_ * D_ * 2;       // 47,185,920
    const size_t QKV  = (size_t)BS_ * 3 * D_ * 4;   // 283,115,520
    const size_t AOB  = (size_t)BS_ * D_ * 2;
    const size_t LAST = (size_t)B_ * D_ * 2;
    const size_t HB   = (size_t)B_ * 192 * 2;

    size_t o = 0;
    float*  xf   = (float*)(ws + o);  o += XF;
    __bf16* xb   = (__bf16*)(ws + o); o += XB;
    char*   qkvR = ws + o;            o += QKV;
    __bf16* aob  = (__bf16*)(ws + o); o += AOB;
    __bf16* lastb= (__bf16*)(ws + o); o += LAST;
    __bf16* hb   = (__bf16*)(ws + o); o += HB;
    __bf16* w_inp = (__bf16*)(ws + o); o += (size_t)288 * 96 * 2;
    __bf16* w_out = (__bf16*)(ws + o); o += (size_t)96 * 96 * 2;
    __bf16* w_ff1 = (__bf16*)(ws + o); o += (size_t)192 * 96 * 2;
    __bf16* w_ff2 = (__bf16*)(ws + o); o += (size_t)96 * 192 * 2;
    __bf16* w_p1  = (__bf16*)(ws + o); o += (size_t)192 * 96 * 2;
    __bf16* w_p2  = (__bf16*)(ws + o); o += (size_t)L_ * 192 * 2;

    // qkv region sub-reuse (lifetimes are disjoint):
    float*  qkvf = (float*)qkvR;                    // [BS,288] f32, live through attn
    float*  ao2f = (float*)qkvR;                    // [BS,96]  f32, after attn
    __bf16* ff1b = (__bf16*)(qkvR + XF);            // [BS,192] bf16, after ln1
    float*  ff2f = (float*)(qkvR + 2 * XF);         // [BS,96]  f32

    // ---- 0. weight conversions to bf16 ----
    launch_cvt(in_proj_w,  w_inp, 288 * 96, stream);
    launch_cvt(out_proj_w, w_out, 96 * 96, stream);
    launch_cvt(ff_w1,      w_ff1, 192 * 96, stream);
    launch_cvt(ff_w2,      w_ff2, 96 * 192, stream);
    launch_cvt(pred_w1,    w_p1,  192 * 96, stream);
    launch_cvt(pred_w2,    w_p2,  L_ * 192, stream);

    // ---- 1. embed + temporal + LN + PE ----
    embed_kernel<<<BS_ / 8, 256, 0, stream>>>(
        loc_seq, user_seq, weekday, start_min, dur, diff,
        loc_emb_w, user_emb_w, temporal_w, temporal_b,
        in_norm_g, in_norm_b, pe, xf, xb, BS_);

    // ---- 2. QKV projection:  [BS,96] x [96,288] ----
    launch_gemm<96, 0, 0>(xb, w_inp, in_proj_b, qkvf, nullptr, BS_, 288, stream);

    // ---- 3. attention ----
    attn_kernel<<<B_ * H_, 64, 0, stream>>>(qkvf, mask, aob);

    // ---- 4. out_proj:  [BS,96] x [96,96] ----
    launch_gemm<96, 0, 0>(aob, w_out, out_proj_b, ao2f, nullptr, BS_, 96, stream);

    // ---- 5. x1 = LN(x + ao)  (in-place into xf/xb) ----
    ln_res_kernel<<<BS_ / 8, 256, 0, stream>>>(xf, ao2f, norm1_g, norm1_b, xf, xb, BS_);

    // ---- 6. FFN: gelu([BS,96]x[96,192]) -> [BS,192]x[192,96] ----
    launch_gemm<96, 1, 0>(xb, w_ff1, ff_b1, nullptr, ff1b, BS_, 192, stream);
    launch_gemm<192, 0, 0>(ff1b, w_ff2, ff_b2, ff2f, nullptr, BS_, 96, stream);

    // ---- 7. x2 = LN(x1 + ff)  (in-place into xf) ----
    ln_res_kernel<<<BS_ / 8, 256, 0, stream>>>(xf, ff2f, norm2_g, norm2_b, xf, nullptr, BS_);

    // ---- 8. gather last valid token ----
    last_kernel<<<(B_ * D_ + 255) / 256, 256, 0, stream>>>(xf, mask, lastb);

    // ---- 9. prediction head: gelu([B,96]x[96,192]) then [B,192]x[192,20000] ----
    launch_gemm<96, 1, 0>(lastb, w_p1, pred_b1, nullptr, hb, B_, 192, stream);
    // Dominant GEMM + 327MB streamed output: non-temporal stores keep pred_w2 in L2.
    launch_gemm<192, 0, 1>(hb, w_p2, pred_b2, out, nullptr, B_, L_, stream);

    // ---- 10. recency-weighted history add (row-private, deterministic) ----
    hist_kernel<<<(B_ + 255) / 256, 256, 0, stream>>>(loc_seq, mask, boost, decay, out);
}
